// Seq2SeqWithAttention_55422257987912
// MI455X (gfx1250) — matrix operand order repaired and checked by reading, not compiled, not run on hardware
//
#include <hip/hip_runtime.h>
#include <hip/hip_bf16.h>
#include <cstddef>

// Problem dims
#define B_   32
#define S_   64
#define V_   96
#define E_   256
#define H_   512
#define L_   3
#define DIN_ 1024     // 2*H
#define G4H_ 2048     // 4*H
#define XCAT_ 1280    // E + 2H

typedef __attribute__((ext_vector_type(16))) __bf16        v16bf;
typedef __attribute__((ext_vector_type(8)))  float         v8f;
typedef __attribute__((ext_vector_type(4)))  unsigned int  u32x4;

union Frag { v16bf v; u32x4 u[2]; };

__device__ __forceinline__ unsigned short f2bf(float f) {
  unsigned u = __float_as_uint(f);
  unsigned r = (u + 0x7FFFu + ((u >> 16) & 1u)) >> 16;
  return (unsigned short)r;
}
__device__ __forceinline__ float sigm_(float x) { return 1.f / (1.f + __expf(-x)); }

// ---------------------------------------------------------------------------
// fp32 -> bf16 weight conversion
// ---------------------------------------------------------------------------
__global__ void cvt_bf16_kernel(const float* __restrict__ s,
                                unsigned short* __restrict__ d, int n) {
  int i = blockIdx.x * 256 + threadIdx.x;
  if (i < n) d[i] = f2bf(s[i]);
}

// Build encoder layer-0 input: xs0[(s*B+b)*DIN + d] = pad(embedding[src[b,s]], DIN)
__global__ void embed_src_kernel(const int* __restrict__ src,
                                 const float* __restrict__ emb,
                                 unsigned short* __restrict__ xs0) {
  int i = blockIdx.x * 256 + threadIdx.x;
  if (i >= S_ * B_ * DIN_) return;
  int d  = i & (DIN_ - 1);
  int sb = i >> 10;          // / DIN_
  int b  = sb & (B_ - 1);
  int s  = sb >> 5;          // / B_
  unsigned short v = 0;
  if (d < E_) { int tok = src[b * S_ + s]; v = f2bf(emb[tok * E_ + d]); }
  xs0[i] = v;
}

__global__ void enc_state_zero_kernel(float* __restrict__ c,
                                      unsigned short* __restrict__ hbf) {
  int i = blockIdx.x * 256 + threadIdx.x;
  if (i < 2 * B_ * H_) { c[i] = 0.f; hbf[i] = 0; }
}

// dec_h[l] = hf+hb ; dec_c[l] = cf+cb ; also bf16 copy of dec_h
__global__ void enc_reduce_kernel(const float* __restrict__ hf32,
                                  const float* __restrict__ c,
                                  float* __restrict__ dh,
                                  unsigned short* __restrict__ dhbf,
                                  float* __restrict__ dc) {
  int i = blockIdx.x * 256 + threadIdx.x;
  if (i >= B_ * H_) return;
  float h  = hf32[i] + hf32[B_ * H_ + i];
  float cc = c[i]    + c[B_ * H_ + i];
  dh[i] = h; dhbf[i] = f2bf(h); dc[i] = cc;
}

__global__ void out_zero_kernel(float* __restrict__ out) {
  int i = blockIdx.x * 256 + threadIdx.x;
  if (i < B_ * V_) out[(size_t)(i / V_) * (S_ * V_) + (i % V_)] = 0.f;
}

// ---------------------------------------------------------------------------
// Fused LSTM gate-GEMM + cell.  Block = 256 threads = 8 waves.
// Wave w: gate g = w>>1 (0..3), m-tile m = w&1.  blockIdx.x = jt (hidden tile).
// gates(B,4H) = x(B,Kx) @ Wih^T + h(B,H) @ Whh^T + bih + bhh, then cell in LDS.
// ---------------------------------------------------------------------------
__device__ void lstm_core(const unsigned short* __restrict__ x, int Kx,
                          const unsigned short* __restrict__ Wih,   // rows stride DIN_
                          const unsigned short* __restrict__ Whh,   // rows stride H_
                          const float* __restrict__ bih,
                          const float* __restrict__ bhh,
                          const unsigned short* __restrict__ h_in,  // (B,H) bf16
                          const float* __restrict__ c_in,
                          float* __restrict__ c_out,
                          float* __restrict__ h_f32,
                          unsigned short* __restrict__ h_bf,
                          unsigned short* __restrict__ y, int ystride,
                          float* __restrict__ yf32, int yfstride,
                          int jt) {
  __shared__ float gl[4][2][16][16];   // [gate][mtile][row][col]
  const int tid  = threadIdx.x;
  const int wave = tid >> 5;
  const int lane = tid & 31;
  const int g    = wave >> 1;
  const int m    = wave & 1;
  const int ln15 = lane & 15;
  const int kh   = lane >> 4;

  const int arow = m * 16 + ln15;            // batch row
  const int wrow = g * H_ + jt * 16 + ln15;  // gate column / weight row

  v8f acc = {0.f, 0.f, 0.f, 0.f, 0.f, 0.f, 0.f, 0.f};

  // x @ Wih^T
  {
    const unsigned short* xr = x   + (size_t)arow * Kx;
    const unsigned short* wr = Wih + (size_t)wrow * DIN_;
    for (int k0 = 0; k0 < Kx; k0 += 32) {
      Frag a, b;
      a.u[0] = *(const u32x4*)(xr + k0 + kh * 8);
      a.u[1] = *(const u32x4*)(xr + k0 + 16 + kh * 8);
      b.u[0] = *(const u32x4*)(wr + k0 + kh * 8);
      b.u[1] = *(const u32x4*)(wr + k0 + 16 + kh * 8);
      acc = __builtin_amdgcn_wmma_f32_16x16x32_bf16(false, a.v, false, b.v,
                                                    (short)0, acc, false, false);
    }
  }
  // h @ Whh^T
  {
    const unsigned short* hr = h_in + (size_t)arow * H_;
    const unsigned short* wr = Whh  + (size_t)wrow * H_;
    for (int k0 = 0; k0 < H_; k0 += 32) {
      Frag a, b;
      a.u[0] = *(const u32x4*)(hr + k0 + kh * 8);
      a.u[1] = *(const u32x4*)(hr + k0 + 16 + kh * 8);
      b.u[0] = *(const u32x4*)(wr + k0 + kh * 8);
      b.u[1] = *(const u32x4*)(wr + k0 + 16 + kh * 8);
      acc = __builtin_amdgcn_wmma_f32_16x16x32_bf16(false, a.v, false, b.v,
                                                    (short)0, acc, false, false);
    }
  }

  const float bias = bih[wrow] + bhh[wrow];
#pragma unroll
  for (int r = 0; r < 8; ++r)
    gl[g][m][r + 8 * kh][ln15] = acc[r] + bias;
  __syncthreads();

  // LSTM cell: 2*16*16 = 512 elements, 256 threads
  for (int e = tid; e < 512; e += 256) {
    int mm = e >> 8, rr = (e >> 4) & 15, cc = e & 15;
    int b = mm * 16 + rr;
    int j = jt * 16 + cc;
    float gi = gl[0][mm][rr][cc];
    float gf = gl[1][mm][rr][cc];
    float gg = gl[2][mm][rr][cc];
    float go = gl[3][mm][rr][cc];
    float cp = c_in[(size_t)b * H_ + j];
    float c2 = sigm_(gf) * cp + sigm_(gi) * tanhf(gg);
    float h2 = sigm_(go) * tanhf(c2);
    c_out[(size_t)b * H_ + j] = c2;
    h_f32[(size_t)b * H_ + j] = h2;
    unsigned short hb = f2bf(h2);
    h_bf[(size_t)b * H_ + j] = hb;
    if (y)    y[(size_t)b * ystride + j]     = hb;
    if (yf32) yf32[(size_t)b * yfstride + j] = h2;
  }
}

// Encoder step: grid (32 jt, 2 dir), block 256.
__global__ void lstm_step_enc_kernel(const unsigned short* __restrict__ xs,
                                     unsigned short* __restrict__ ys,
                                     float* __restrict__ yf32,  // null except last layer
                                     const unsigned short* __restrict__ Wih,
                                     const unsigned short* __restrict__ Whh,
                                     const float* __restrict__ bih,
                                     const float* __restrict__ bhh,
                                     const unsigned short* __restrict__ h_in,
                                     unsigned short* __restrict__ h_out,
                                     float* __restrict__ h_f32,
                                     float* __restrict__ c, int t) {
  const int dir  = blockIdx.y;
  const int jt   = blockIdx.x;
  const int teff = dir ? (S_ - 1 - t) : t;
  const unsigned short* x = xs + (size_t)teff * B_ * DIN_;
  unsigned short*       y = ys + (size_t)teff * B_ * DIN_ + dir * H_;
  float* yf = yf32 ? (yf32 + (size_t)teff * B_ * DIN_ + dir * H_) : nullptr;
  const size_t so = (size_t)dir * B_ * H_;
  lstm_core(x, DIN_,
            Wih + (size_t)dir * G4H_ * DIN_, Whh + (size_t)dir * G4H_ * H_,
            bih + dir * G4H_, bhh + dir * G4H_,
            h_in + so, c + so, c + (size_t)0 + so, h_f32 + so, h_out + so,
            y, DIN_, yf, DIN_, jt);
}

// Decoder step layer: grid (32,1), block 256.
__global__ void lstm_step_dec_kernel(const unsigned short* __restrict__ x, int Kx,
                                     const unsigned short* __restrict__ Wih,
                                     const unsigned short* __restrict__ Whh,
                                     const float* __restrict__ bih,
                                     const float* __restrict__ bhh,
                                     const unsigned short* __restrict__ h_in,
                                     unsigned short* __restrict__ h_out,
                                     float* __restrict__ h_f32,
                                     float* __restrict__ c) {
  lstm_core(x, Kx, Wih, Whh, bih, bhh, h_in, c, c, h_f32, h_out,
            nullptr, 0, nullptr, 0, blockIdx.x);
}

// ---------------------------------------------------------------------------
// Generic WMMA GEMM: C(M,N) = A(M,K)bf16 @ Bw(N,K)bf16^T + bias
// block = 256 (8 waves), one 16x16 tile per wave.
// ---------------------------------------------------------------------------
__global__ void gemm_bf16_kernel(const unsigned short* __restrict__ A, int lda,
                                 const unsigned short* __restrict__ Bw, int ldb,
                                 const float* __restrict__ bias,
                                 float* __restrict__ Cf,
                                 unsigned short* __restrict__ Cb, int ldc,
                                 int Mt, int Nt, int K) {
  const int wave = threadIdx.x >> 5;
  const int lane = threadIdx.x & 31;
  const int tile = blockIdx.x * 8 + wave;
  if (tile >= Mt * Nt) return;
  const int mt = tile / Nt, nt = tile % Nt;
  const int ln15 = lane & 15, kh = lane >> 4;

  const unsigned short* ar = A  + (size_t)(mt * 16 + ln15) * lda;
  const unsigned short* br = Bw + (size_t)(nt * 16 + ln15) * ldb;

  v8f acc = {0.f, 0.f, 0.f, 0.f, 0.f, 0.f, 0.f, 0.f};
  for (int k0 = 0; k0 < K; k0 += 32) {
    Frag a, b;
    a.u[0] = *(const u32x4*)(ar + k0 + kh * 8);
    a.u[1] = *(const u32x4*)(ar + k0 + 16 + kh * 8);
    b.u[0] = *(const u32x4*)(br + k0 + kh * 8);
    b.u[1] = *(const u32x4*)(br + k0 + 16 + kh * 8);
    acc = __builtin_amdgcn_wmma_f32_16x16x32_bf16(false, a.v, false, b.v,
                                                  (short)0, acc, false, false);
  }
  const int col = nt * 16 + ln15;
  const float bs = bias ? bias[col] : 0.f;
#pragma unroll
  for (int r = 0; r < 8; ++r) {
    const int row = mt * 16 + r + 8 * kh;
    float v = acc[r] + bs;
    if (Cf) Cf[(size_t)row * ldc + col] = v;
    if (Cb) Cb[(size_t)row * ldc + col] = f2bf(v);
  }
}

// ---------------------------------------------------------------------------
// Attention scores + softmax + context + xcat assembly.  grid = B, block = 256.
// scores[b,s] = v . tanh(q[b] + enc_att[s,b]) ; attn = softmax_s ;
// context[b,d] = sum_s attn * enc_out[s,b,d] ; xcat = [emb(trg[b,step]) | ctx]
// ---------------------------------------------------------------------------
__global__ void attn_ctx_kernel(const float* __restrict__ q,
                                const float* __restrict__ enc_att,  // (S,B,H)
                                const float* __restrict__ enc_out,  // (S,B,2H) f32
                                const float* __restrict__ attn_v,
                                const float* __restrict__ emb,
                                const int* __restrict__ trg, int step,
                                unsigned short* __restrict__ xcat) {
  const int b = blockIdx.x;
  const int tid = threadIdx.x;
  __shared__ float sc[S_];
  __shared__ float part[256];

  const int s = tid & 63, ch = tid >> 6;     // 4 threads per s, 128 dims each
  const float* qa = q + (size_t)b * H_;
  const float* ea = enc_att + ((size_t)s * B_ + b) * H_;
  float p = 0.f;
  for (int d = ch * 128; d < ch * 128 + 128; ++d)
    p += attn_v[d] * tanhf(qa[d] + ea[d]);
  part[tid] = p;
  __syncthreads();
  if (ch == 0) sc[s] = part[s] + part[64 + s] + part[128 + s] + part[192 + s];
  __syncthreads();
  if (tid == 0) {
    float mx = sc[0];
    for (int i = 1; i < S_; ++i) mx = fmaxf(mx, sc[i]);
    float sum = 0.f;
    for (int i = 0; i < S_; ++i) { float e = __expf(sc[i] - mx); sc[i] = e; sum += e; }
    float inv = 1.f / sum;
    for (int i = 0; i < S_; ++i) sc[i] *= inv;
  }
  __syncthreads();

  const int tok = trg[b * S_ + step];
  for (int d = tid; d < E_; d += 256)
    xcat[(size_t)b * XCAT_ + d] = f2bf(emb[tok * E_ + d]);
  for (int d = tid; d < DIN_; d += 256) {
    float a = 0.f;
    for (int s2 = 0; s2 < S_; ++s2)
      a += sc[s2] * enc_out[((size_t)s2 * B_ + b) * DIN_ + d];
    xcat[(size_t)b * XCAT_ + E_ + d] = f2bf(a);
  }
}

// ---------------------------------------------------------------------------
extern "C" void kernel_launch(void* const* d_in, const int* in_sizes, int n_in,
                              void* d_out, int out_size, void* d_ws, size_t ws_size,
                              hipStream_t stream) {
  (void)in_sizes; (void)n_in; (void)out_size; (void)ws_size;
  const int*   src     = (const int*)d_in[0];
  const int*   trg     = (const int*)d_in[1];
  const float* emb     = (const float*)d_in[2];
  const float* encWih  = (const float*)d_in[3];
  const float* encWhh  = (const float*)d_in[4];
  const float* encBih  = (const float*)d_in[5];
  const float* encBhh  = (const float*)d_in[6];
  const float* decWih  = (const float*)d_in[7];
  const float* decWhh  = (const float*)d_in[8];
  const float* decBih  = (const float*)d_in[9];
  const float* decBhh  = (const float*)d_in[10];
  const float* attnW   = (const float*)d_in[11];
  const float* attnB   = (const float*)d_in[12];
  const float* attnV   = (const float*)d_in[13];
  const float* projW   = (const float*)d_in[14];
  const float* projB   = (const float*)d_in[15];
  const float* fcW     = (const float*)d_in[16];
  const float* fcB     = (const float*)d_in[17];
  float* out = (float*)d_out;

  // ---- workspace partitioning (deterministic) ----
  char* base = (char*)d_ws;
  size_t off = 0;
  auto take = [&](size_t bytes) -> char* {
    char* p = base + off;
    off = (off + bytes + 255) & ~(size_t)255;
    return p;
  };
  const size_t nEncWih = (size_t)L_ * 2 * G4H_ * DIN_;   // 12.58M
  const size_t nEncWhh = (size_t)L_ * 2 * G4H_ * H_;     // 6.29M
  const size_t nDecWih = (size_t)L_ * G4H_ * DIN_;       // 6.29M
  const size_t nDecWhh = (size_t)L_ * G4H_ * H_;         // 3.15M
  const size_t nAttnW  = (size_t)H_ * 3 * H_;            // 0.79M
  const size_t nProjW  = (size_t)DIN_ * XCAT_;           // 1.31M
  const size_t nFcW    = (size_t)V_ * H_;

  unsigned short* bEncWih = (unsigned short*)take(nEncWih * 2);
  unsigned short* bEncWhh = (unsigned short*)take(nEncWhh * 2);
  unsigned short* bDecWih = (unsigned short*)take(nDecWih * 2);
  unsigned short* bDecWhh = (unsigned short*)take(nDecWhh * 2);
  unsigned short* bAttnW  = (unsigned short*)take(nAttnW * 2);
  unsigned short* bProjW  = (unsigned short*)take(nProjW * 2);
  unsigned short* bFcW    = (unsigned short*)take(nFcW * 2);

  unsigned short* xsA  = (unsigned short*)take((size_t)S_ * B_ * DIN_ * 2);
  unsigned short* xsB  = (unsigned short*)take((size_t)S_ * B_ * DIN_ * 2);
  float* encOutF       = (float*)take((size_t)S_ * B_ * DIN_ * 4);
  float* encAtt        = (float*)take((size_t)S_ * B_ * H_ * 4);
  unsigned short* ehA  = (unsigned short*)take((size_t)2 * B_ * H_ * 2);
  unsigned short* ehB  = (unsigned short*)take((size_t)2 * B_ * H_ * 2);
  float* ehF           = (float*)take((size_t)2 * B_ * H_ * 4);
  float* ec            = (float*)take((size_t)2 * B_ * H_ * 4);
  float* dh            = (float*)take((size_t)L_ * B_ * H_ * 4);
  unsigned short* dhbA = (unsigned short*)take((size_t)L_ * B_ * H_ * 2);
  unsigned short* dhbB = (unsigned short*)take((size_t)L_ * B_ * H_ * 2);
  float* dc            = (float*)take((size_t)L_ * B_ * H_ * 4);
  float* qbuf          = (float*)take((size_t)B_ * H_ * 4);
  unsigned short* xcat = (unsigned short*)take((size_t)B_ * XCAT_ * 2);
  unsigned short* x0   = (unsigned short*)take((size_t)B_ * DIN_ * 2);

  auto cvt = [&](const float* s, unsigned short* d, size_t n) {
    cvt_bf16_kernel<<<(unsigned)((n + 255) / 256), 256, 0, stream>>>(s, d, (int)n);
  };

  // ---- weight conversion (all bf16 weights together fit in the 192MB L2) ----
  cvt(encWih, bEncWih, nEncWih);
  cvt(encWhh, bEncWhh, nEncWhh);
  cvt(decWih, bDecWih, nDecWih);
  cvt(decWhh, bDecWhh, nDecWhh);
  cvt(attnW,  bAttnW,  nAttnW);
  cvt(projW,  bProjW,  nProjW);
  cvt(fcW,    bFcW,    nFcW);

  // ---- encoder layer-0 input ----
  embed_src_kernel<<<(S_ * B_ * DIN_) / 256, 256, 0, stream>>>(src, emb, xsA);

  // ---- encoder: 3 bidirectional layers, 64 sequential steps each ----
  unsigned short* xs_in  = xsA;
  unsigned short* xs_out = xsB;
  for (int l = 0; l < L_; ++l) {
    enc_state_zero_kernel<<<(2 * B_ * H_) / 256, 256, 0, stream>>>(ec, ehA);
    const unsigned short* Wih_l = bEncWih + (size_t)l * 2 * G4H_ * DIN_;
    const unsigned short* Whh_l = bEncWhh + (size_t)l * 2 * G4H_ * H_;
    const float* bih_l = encBih + (size_t)l * 2 * G4H_;
    const float* bhh_l = encBhh + (size_t)l * 2 * G4H_;
    unsigned short* hin  = ehA;
    unsigned short* hout = ehB;
    float* yf32 = (l == L_ - 1) ? encOutF : nullptr;
    for (int t = 0; t < S_; ++t) {
      lstm_step_enc_kernel<<<dim3(32, 2), 256, 0, stream>>>(
          xs_in, xs_out, yf32, Wih_l, Whh_l, bih_l, bhh_l, hin, hout, ehF, ec, t);
      unsigned short* tmp = hin; hin = hout; hout = tmp;
    }
    enc_reduce_kernel<<<(B_ * H_) / 256, 256, 0, stream>>>(
        ehF, ec, dh + (size_t)l * B_ * H_, dhbA + (size_t)l * B_ * H_,
        dc + (size_t)l * B_ * H_);
    unsigned short* tmp = xs_in; xs_in = xs_out; xs_out = tmp;
  }
  // xs_in now holds enc_out bf16 (S,B,2H); encOutF holds it in f32.
  const unsigned short* encOutBf = xs_in;

  // ---- precompute enc_att = enc_out @ We^T + attn_b   (We = attn_W[:, H:3H]) ----
  {
    const int Mt = (S_ * B_) / 16, Nt = H_ / 16;   // 128 x 32
    gemm_bf16_kernel<<<(Mt * Nt + 7) / 8, 256, 0, stream>>>(
        encOutBf, DIN_, bAttnW + H_, 3 * H_, attnB, encAtt, nullptr, H_, Mt, Nt, DIN_);
  }

  // outputs[:,0,:] = 0
  out_zero_kernel<<<(B_ * V_ + 255) / 256, 256, 0, stream>>>(out);

  // ---- decoder: 63 sequential steps ----
  for (int step = 0; step < S_ - 1; ++step) {
    unsigned short* hbin  = (step & 1) ? dhbB : dhbA;
    unsigned short* hbout = (step & 1) ? dhbA : dhbB;

    // q = h_top @ Wh^T   (Wh = attn_W[:, :H])
    {
      const int Mt = B_ / 16, Nt = H_ / 16;   // 2 x 32
      gemm_bf16_kernel<<<(Mt * Nt + 7) / 8, 256, 0, stream>>>(
          hbin + (size_t)(L_ - 1) * B_ * H_, H_, bAttnW, 3 * H_, nullptr,
          qbuf, nullptr, H_, Mt, Nt, H_);
    }
    // attention scores + softmax + context + xcat
    attn_ctx_kernel<<<B_, 256, 0, stream>>>(qbuf, encAtt, encOutF, attnV, emb,
                                            trg, step, xcat);
    // x0 = xcat @ proj_W^T + proj_b   -> bf16 (B, 1024)
    {
      const int Mt = B_ / 16, Nt = DIN_ / 16;  // 2 x 64
      gemm_bf16_kernel<<<(Mt * Nt + 7) / 8, 256, 0, stream>>>(
          xcat, XCAT_, bProjW, XCAT_, projB, nullptr, x0, DIN_, Mt, Nt, XCAT_);
    }
    // 3 decoder LSTM layers (layer l>0 input = layer l-1 hidden, zero-padded == Kx=512)
    for (int l = 0; l < L_; ++l) {
      const unsigned short* xl = (l == 0) ? x0 : (hbout + (size_t)(l - 1) * B_ * H_);
      const int Kx = (l == 0) ? DIN_ : H_;
      lstm_step_dec_kernel<<<dim3(32, 1), 256, 0, stream>>>(
          xl, Kx,
          bDecWih + (size_t)l * G4H_ * DIN_, bDecWhh + (size_t)l * G4H_ * H_,
          decBih + (size_t)l * G4H_, decBhh + (size_t)l * G4H_,
          hbin + (size_t)l * B_ * H_, hbout + (size_t)l * B_ * H_,
          dh + (size_t)l * B_ * H_, dc + (size_t)l * B_ * H_);
    }
    // out[:, step+1, :] = h_top @ fc_W^T + fc_b
    {
      const int Mt = B_ / 16, Nt = V_ / 16;    // 2 x 6
      gemm_bf16_kernel<<<(Mt * Nt + 7) / 8, 256, 0, stream>>>(
          hbout + (size_t)(L_ - 1) * B_ * H_, H_, bFcW, H_, fcB,
          out + (size_t)(step + 1) * V_, nullptr, S_ * V_, Mt, Nt, H_);
    }
  }
}
